// GroupedExperts_66554813219384
// MI455X (gfx1250) — matrix operand order repaired and checked by reading, not compiled, not run on hardware
//
#include <hip/hip_runtime.h>
#include <hip/hip_bf16.h>
#include <type_traits>

namespace {
constexpr int kB     = 2;
constexpr int kT     = 4096;
constexpr int kDIM   = 1024;
constexpr int kHID   = 512;
constexpr int kE     = 16;
constexpr int kTileM = 64;                 // tokens per workgroup

constexpr int kXStride = kDIM + 8;         // LDS row stride (bf16) -> 4 dword-banks skew
constexpr int kHStride = kHID + 8;         // LDS row stride (bf16) -> 4 dword-banks skew

constexpr size_t kXElems = (size_t)kB * kT * kDIM;      // 8388608
constexpr size_t kWElems = (size_t)kE * kHID * kDIM;    // 8388608 (w1/w2/w3 all same size)
}

typedef __attribute__((ext_vector_type(16))) __bf16       bf16x16;
typedef __attribute__((ext_vector_type(16))) short        s16x16;
typedef __attribute__((ext_vector_type(8)))  float        v8f;
typedef __attribute__((ext_vector_type(8)))  unsigned int u32x8;
typedef __attribute__((ext_vector_type(4)))  int          v4i;

// fp32 -> bf16 raw bits, round-to-nearest-even
__device__ __forceinline__ unsigned short f2bf(float f) {
  unsigned int u = __float_as_uint(f);
  u += 0x7FFFu + ((u >> 16) & 1u);
  return (unsigned short)(u >> 16);
}

__device__ __forceinline__ s16x16 pack16(float4 a0, float4 a1, float4 a2, float4 a3) {
  s16x16 s;
  s[0]  = (short)f2bf(a0.x); s[1]  = (short)f2bf(a0.y); s[2]  = (short)f2bf(a0.z); s[3]  = (short)f2bf(a0.w);
  s[4]  = (short)f2bf(a1.x); s[5]  = (short)f2bf(a1.y); s[6]  = (short)f2bf(a1.z); s[7]  = (short)f2bf(a1.w);
  s[8]  = (short)f2bf(a2.x); s[9]  = (short)f2bf(a2.y); s[10] = (short)f2bf(a2.z); s[11] = (short)f2bf(a2.w);
  s[12] = (short)f2bf(a3.x); s[13] = (short)f2bf(a3.y); s[14] = (short)f2bf(a3.z); s[15] = (short)f2bf(a3.w);
  return s;
}

__device__ __forceinline__ bf16x16 two_uint4_frag(uint4 u0, uint4 u1) {
  u32x8 u;
  u[0] = u0.x; u[1] = u0.y; u[2] = u0.z; u[3] = u0.w;
  u[4] = u1.x; u[5] = u1.y; u[6] = u1.z; u[7] = u1.w;
  return __builtin_bit_cast(bf16x16, u);
}

// ---- A fragment (16x32 bf16), lane%16 = row M.
// elem e<8 -> K = k0 + hlf*8 + e ; elem e>=8 -> K = k0 + 16 + hlf*8 + (e-8)
__device__ __forceinline__ bf16x16 load_a_g(const float* __restrict__ rowptr, int k0, int hlf) {
  const float4* p0 = (const float4*)(rowptr + k0 + hlf * 8);
  const float4* p1 = (const float4*)(rowptr + k0 + 16 + hlf * 8);
  return __builtin_bit_cast(bf16x16, pack16(p0[0], p0[1], p1[0], p1[1]));
}
// A fragment from bf16 (LDS or global), 16B-aligned b128 pairs
__device__ __forceinline__ bf16x16 load_a_bf(const unsigned short* rowptr, int k0, int hlf) {
  const uint4* p0 = (const uint4*)(rowptr + k0 + hlf * 8);        // 8 bf16 = 16B
  const uint4* p1 = (const uint4*)(rowptr + k0 + 16 + hlf * 8);
  return two_uint4_frag(*p0, *p1);
}

// ---- B fragment (32x16 bf16) = W^T tile, W row-major [N, K] stride ldk.
// lane%16 = column N-n0 ; elem e<8 -> K = k0 + hlf*8 + e ; elem e>=8 -> K = k0 + 16 + hlf*8 + (e-8)
__device__ __forceinline__ bf16x16 load_b(const float* __restrict__ W, int ldk,
                                          int n0, int k0, int lr, int hlf) {
  const float4* p = (const float4*)(W + (size_t)(n0 + lr) * ldk + k0 + hlf * 8);
  const float4* q = (const float4*)(W + (size_t)(n0 + lr) * ldk + k0 + 16 + hlf * 8);
  return __builtin_bit_cast(bf16x16, pack16(p[0], p[1], q[0], q[1]));
}
__device__ __forceinline__ bf16x16 load_b(const unsigned short* __restrict__ W, int ldk,
                                          int n0, int k0, int lr, int hlf) {
  const uint4* p = (const uint4*)(W + (size_t)(n0 + lr) * ldk + k0 + hlf * 8);
  const uint4* q = (const uint4*)(W + (size_t)(n0 + lr) * ldk + k0 + 16 + hlf * 8);
  return two_uint4_frag(p[0], q[0]);
}

__device__ __forceinline__ v8f wmma_bf16(bf16x16 a, bf16x16 b, v8f c) {
  return __builtin_amdgcn_wmma_f32_16x16x32_bf16(false, a, false, b, (short)0, c, false, false);
}

// ---------------- elementwise fp32 -> bf16 pre-conversion (8 elems/thread)
__global__ __launch_bounds__(256)
void cvt_f32_bf16_kernel(const float* __restrict__ src, unsigned short* __restrict__ dst, int n8) {
  int i = blockIdx.x * blockDim.x + threadIdx.x;
  if (i >= n8) return;
  const float4* p = (const float4*)src + (size_t)i * 2;
  float4 a = p[0], b = p[1];
  uint4 o;
  o.x = (unsigned)f2bf(a.x) | ((unsigned)f2bf(a.y) << 16);
  o.y = (unsigned)f2bf(a.z) | ((unsigned)f2bf(a.w) << 16);
  o.z = (unsigned)f2bf(b.x) | ((unsigned)f2bf(b.y) << 16);
  o.w = (unsigned)f2bf(b.z) | ((unsigned)f2bf(b.w) << 16);
  ((uint4*)dst)[i] = o;
}

// ---------------- grouped-expert SwiGLU MLP, WMMA bf16, templated on source type
template <typename T>
__global__ __launch_bounds__(256)
void moe_swiglu_wmma_kernel(const T* __restrict__ x, const int* __restrict__ counts,
                            const T* __restrict__ w1, const T* __restrict__ w2,
                            const T* __restrict__ w3, float* __restrict__ out) {
  constexpr bool kBF = std::is_same<T, unsigned short>::value;

  __shared__ unsigned short lds_h[kTileM * kHStride];               // ~65 KiB SwiGLU staging
  __shared__ unsigned short lds_x[kBF ? kTileM * kXStride : 8];     // ~129 KiB x-tile (bf16 path)

  const int b    = blockIdx.y;
  const int t0   = blockIdx.x * kTileM;
  const int wave = threadIdx.x >> 5;
  const int lane = threadIdx.x & 31;
  const int hlf  = lane >> 4;
  const int lr   = lane & 15;

  // uniform per-batch-row expert offsets (tokens are contiguous per expert)
  int off[kE + 1];
  off[0] = 0;
#pragma unroll
  for (int e = 0; e < kE; ++e) off[e + 1] = off[e] + counts[b * kE + e];

  // ---- stage the (expert-independent) 64x1024 bf16 x-tile into LDS once per block
  if constexpr (kBF) {
    const unsigned short* xtile = x + ((size_t)b * kT + t0) * kDIM;
#pragma unroll 1
    for (int it = 0; it < 32; ++it) {
      const int idx   = it * 256 + threadIdx.x;   // 16B chunk id, 8192 total
      const int row   = idx >> 7;                 // /128 chunks per row
      const int col16 = idx & 127;
      const unsigned short* gp = xtile + (size_t)row * kDIM + col16 * 8;
      unsigned short*       lp = lds_x + row * kXStride + col16 * 8;
#if __has_builtin(__builtin_amdgcn_global_load_async_to_lds_b128)
      __builtin_amdgcn_global_load_async_to_lds_b128(
          (__attribute__((address_space(1))) v4i*)gp,
          (__attribute__((address_space(3))) v4i*)lp, 0, 0);
#else
      *(uint4*)lp = *(const uint4*)gp;
#endif
    }
#if __has_builtin(__builtin_amdgcn_global_load_async_to_lds_b128)
#if __has_builtin(__builtin_amdgcn_s_wait_asynccnt)
    __builtin_amdgcn_s_wait_asynccnt(0);
#else
    asm volatile("s_wait_asynccnt 0" ::: "memory");
#endif
#endif
    __syncthreads();
  }

  for (int e = 0; e < kE; ++e) {
    const int s  = off[e]     > t0           ? off[e]     : t0;
    const int en = off[e + 1] < t0 + kTileM  ? off[e + 1] : t0 + kTileM;
    if (s >= en) continue;  // uniform across the block -> barriers below are safe

    const T* w1e = w1 + (size_t)e * kHID * kDIM;
    const T* w3e = w3 + (size_t)e * kHID * kDIM;
    const T* w2e = w2 + (size_t)e * kDIM * kHID;

    // ---------- phase 1: h = silu(x @ w1^T) * (x @ w3^T)  -> LDS bf16 [64,512]
#pragma unroll 1
    for (int c = 0; c < 2; ++c) {
      v8f acc1[4][2] = {};
      v8f acc3[4][2] = {};
#pragma unroll 1
      for (int k0 = 0; k0 < kDIM; k0 += 32) {
        bf16x16 a[4];
#pragma unroll
        for (int mt = 0; mt < 4; ++mt) {
          if constexpr (kBF)
            a[mt] = load_a_bf(lds_x + (mt * 16 + lr) * kXStride, k0, hlf);
          else
            a[mt] = load_a_g(x + ((size_t)b * kT + t0 + mt * 16 + lr) * kDIM, k0, hlf);
        }
#pragma unroll
        for (int i = 0; i < 2; ++i) {
          const int n0 = (wave * 4 + c * 2 + i) * 16;
          bf16x16 bw1 = load_b(w1e, kDIM, n0, k0, lr, hlf);
#pragma unroll
          for (int mt = 0; mt < 4; ++mt) acc1[mt][i] = wmma_bf16(a[mt], bw1, acc1[mt][i]);
          bf16x16 bw3 = load_b(w3e, kDIM, n0, k0, lr, hlf);
#pragma unroll
          for (int mt = 0; mt < 4; ++mt) acc3[mt][i] = wmma_bf16(a[mt], bw3, acc3[mt][i]);
        }
      }
      // SwiGLU epilogue: C/D layout -> M = j + 8*hlf (+16*mt), N = lr (+n0)
#pragma unroll
      for (int mt = 0; mt < 4; ++mt)
#pragma unroll
        for (int i = 0; i < 2; ++i) {
          const int n0 = (wave * 4 + c * 2 + i) * 16;
#pragma unroll
          for (int j = 0; j < 8; ++j) {
            const float g   = acc1[mt][i][j];
            const float sig = __builtin_amdgcn_rcpf(1.0f + __expf(-g));  // v_rcp_f32
            const float hv  = g * sig * acc3[mt][i][j];
            lds_h[(mt * 16 + j + 8 * hlf) * kHStride + n0 + lr] = f2bf(hv);
          }
        }
    }
    __syncthreads();

    // ---------- phase 2: y = h @ w2^T  -> [64,1024] fp32 stores (masked per expert slice)
#pragma unroll 1
    for (int c = 0; c < 2; ++c) {
      v8f acc2[4][4] = {};
#pragma unroll 1
      for (int k0 = 0; k0 < kHID; k0 += 32) {
        bf16x16 a[4];
#pragma unroll
        for (int mt = 0; mt < 4; ++mt)
          a[mt] = load_a_bf(lds_h + (mt * 16 + lr) * kHStride, k0, hlf);
#pragma unroll
        for (int i = 0; i < 4; ++i) {
          const int n0 = (wave * 8 + c * 4 + i) * 16;
          bf16x16 bw2 = load_b(w2e, kHID, n0, k0, lr, hlf);
#pragma unroll
          for (int mt = 0; mt < 4; ++mt) acc2[mt][i] = wmma_bf16(a[mt], bw2, acc2[mt][i]);
        }
      }
#pragma unroll
      for (int mt = 0; mt < 4; ++mt)
#pragma unroll
        for (int i = 0; i < 4; ++i) {
          const int n0 = (wave * 8 + c * 4 + i) * 16;
#pragma unroll
          for (int j = 0; j < 8; ++j) {
            const int t = t0 + mt * 16 + j + 8 * hlf;
            if (t >= s && t < en)
              out[((size_t)b * kT + t) * kDIM + n0 + lr] = acc2[mt][i][j];
          }
        }
    }
    __syncthreads();  // lds_h is overwritten if another expert intersects this tile
  }
}

extern "C" void kernel_launch(void* const* d_in, const int* in_sizes, int n_in,
                              void* d_out, int out_size, void* d_ws, size_t ws_size,
                              hipStream_t stream) {
  const float* x      = (const float*)d_in[0];
  const int*   counts = (const int*)  d_in[1];
  const float* w1     = (const float*)d_in[2];
  const float* w2     = (const float*)d_in[3];
  const float* w3     = (const float*)d_in[4];
  float*       out    = (float*)d_out;

  const dim3 grid(kT / kTileM, kB);   // (64, 2) blocks
  const dim3 block(256);              // 8 wave32

  const size_t need = (kXElems + 3 * kWElems) * sizeof(unsigned short);  // 64 MiB
  if (ws_size >= need) {
    // one-time fp32 -> bf16 conversion of activations + weights into workspace
    unsigned short* xb  = (unsigned short*)d_ws;
    unsigned short* w1b = xb  + kXElems;
    unsigned short* w2b = w1b + kWElems;
    unsigned short* w3b = w2b + kWElems;

    const int nx8 = (int)(kXElems / 8), nw8 = (int)(kWElems / 8);
    const dim3 cblk(256);
    hipLaunchKernelGGL(cvt_f32_bf16_kernel, dim3((nx8 + 255) / 256), cblk, 0, stream, x,  xb,  nx8);
    hipLaunchKernelGGL(cvt_f32_bf16_kernel, dim3((nw8 + 255) / 256), cblk, 0, stream, w1, w1b, nw8);
    hipLaunchKernelGGL(cvt_f32_bf16_kernel, dim3((nw8 + 255) / 256), cblk, 0, stream, w2, w2b, nw8);
    hipLaunchKernelGGL(cvt_f32_bf16_kernel, dim3((nw8 + 255) / 256), cblk, 0, stream, w3, w3b, nw8);

    hipLaunchKernelGGL(moe_swiglu_wmma_kernel<unsigned short>, grid, block, 0, stream,
                       xb, counts, w1b, w2b, w3b, out);
  } else {
    // fallback: convert fp32 -> bf16 inline in the GEMM kernel
    hipLaunchKernelGGL(moe_swiglu_wmma_kernel<float>, grid, block, 0, stream,
                       x, counts, w1, w2, w3, out);
  }
}
